// CoattentiveAggregation_88888643158469
// MI455X (gfx1250) — matrix operand order repaired and checked by reading, not compiled, not run on hardware
//
#include <hip/hip_runtime.h>

typedef __attribute__((ext_vector_type(8)))  _Float16 v8h;
typedef __attribute__((ext_vector_type(16))) _Float16 v16h;
typedef __attribute__((ext_vector_type(8)))  float    v8f;

#define N_NODES 20000
#define KNB 32
#define DF  128
#define WPB 2   // waves (=nodes) per block; 20000 % 2 == 0, no tail

struct WaveLDS {
  _Float16 Dm[KNB][DF];   // 8 KB, row-major f16
  _Float16 Qm[KNB][DF];   // 8 KB
  float    E[KNB][KNB];   // 4 KB  exp(L - max)
  float    ci[KNB];       // 1/colsum
  float    ri[KNB];       // 1/rowsum
  float    wv[KNB];       // w = colmean(AS)
  float    tv[KNB];       // w/r
  float    uv[KNB];       // u = E^T (w/r)
  float    Hm[3 * DF];    // [mean_k Qm, w^T Dm, u^T Qm]
};

__global__ __launch_bounds__(WPB * 32)
void coatt_kernel(const float* __restrict__ feat,
                  const int*   __restrict__ sim_idx,
                  const int*   __restrict__ cor_idx,
                  float*       __restrict__ out)
{
  __shared__ WaveLDS smem[WPB];
  const int wave = threadIdx.x >> 5;
  const int lane = threadIdx.x & 31;
  const int node = blockIdx.x * WPB + wave;
  WaveLDS& W = smem[wave];

  const int lm = lane & 15;   // tile row/col within 16
  const int g  = lane >> 4;   // half-wave group

  // ---------- gather: one coalesced 512B feat row per neighbor ----------
  const int* srow = sim_idx + (size_t)node * KNB;
  const int* qrow = cor_idx + (size_t)node * KNB;
  __builtin_prefetch(srow, 0, 0);
  __builtin_prefetch(qrow, 0, 0);
  for (int r = 0; r < KNB; ++r) {
    const int si = srow[r];                 // uniform -> scalar load
    const int qi = qrow[r];
    float4 a = ((const float4*)(feat + (size_t)si * DF))[lane];
    float4 b = ((const float4*)(feat + (size_t)qi * DF))[lane];
    const int d0 = lane * 4;
    W.Dm[r][d0+0] = (_Float16)a.x; W.Dm[r][d0+1] = (_Float16)a.y;
    W.Dm[r][d0+2] = (_Float16)a.z; W.Dm[r][d0+3] = (_Float16)a.w;
    W.Qm[r][d0+0] = (_Float16)b.x; W.Qm[r][d0+1] = (_Float16)b.y;
    W.Qm[r][d0+2] = (_Float16)b.z; W.Qm[r][d0+3] = (_Float16)b.w;
  }
  __syncthreads();

  // ---------- L = Dm * Qm^T  (M=N=32, K=128) via v_wmma_f32_16x16x32_f16 ----------
  v8f acc[2][2] = {};
  #pragma unroll
  for (int k0 = 0; k0 < DF; k0 += 32) {
    #pragma unroll
    for (int mi = 0; mi < 2; ++mi) {
      // A fragment (16x32 f16): lane holds row mi*16+lm; halfs 0..7 -> K=k0+g*8.. ,
      // halfs 8..15 -> K=k0+16+g*8..
      const int row = mi * 16 + lm;
      v8h alo = *(const v8h*)&W.Dm[row][k0 + g * 8];
      v8h ahi = *(const v8h*)&W.Dm[row][k0 + 16 + g * 8];
      v16h a;
      #pragma unroll
      for (int i = 0; i < 8; ++i) { a[i] = alo[i]; a[i + 8] = ahi[i]; }
      #pragma unroll
      for (int ni = 0; ni < 2; ++ni) {
        // B fragment (32x16 f16): B[k][n] = Qm[n][k]; lane holds col ni*16+lm,
        // halfs 0..7 -> K=k0+g*8.., halfs 8..15 -> K=k0+16+g*8.. (same as A)
        const int col = ni * 16 + lm;
        v8h blo = *(const v8h*)&W.Qm[col][k0 + g * 8];
        v8h bhi = *(const v8h*)&W.Qm[col][k0 + 16 + g * 8];
        v16h b;
        #pragma unroll
        for (int i = 0; i < 8; ++i) { b[i] = blo[i]; b[i + 8] = bhi[i]; }
        acc[mi][ni] = __builtin_amdgcn_wmma_f32_16x16x32_f16(
            false, a, false, b, (short)0, acc[mi][ni], false, false);
      }
    }
  }

  // ---------- global max (softmax shift; cancels exactly in both softmaxes) ----------
  float mx = -3.0e38f;
  #pragma unroll
  for (int mi = 0; mi < 2; ++mi)
    #pragma unroll
    for (int ni = 0; ni < 2; ++ni)
      #pragma unroll
      for (int t = 0; t < 8; ++t) mx = fmaxf(mx, acc[mi][ni][t]);
  #pragma unroll
  for (int off = 16; off > 0; off >>= 1)
    mx = fmaxf(mx, __shfl_xor(mx, off, 32));

  // ---------- E = exp(L - mx) to LDS (C/D layout: M = mi*16 + g*8 + t, N = ni*16+lm) ----------
  #pragma unroll
  for (int mi = 0; mi < 2; ++mi)
    #pragma unroll
    for (int ni = 0; ni < 2; ++ni)
      #pragma unroll
      for (int t = 0; t < 8; ++t)
        W.E[mi * 16 + g * 8 + t][ni * 16 + lm] = __expf(acc[mi][ni][t] - mx);
  __syncthreads();

  // ---------- row/col sums, w, u (lane = one of 32 rows/cols) ----------
  float rsum = 0.f, csum = 0.f;
  for (int j = 0; j < KNB; ++j) {
    rsum += W.E[lane][j];
    csum += W.E[j][lane];
  }
  W.ri[lane] = 1.0f / rsum;
  W.ci[lane] = 1.0f / csum;
  __syncthreads();

  float wacc = 0.f;
  for (int k = 0; k < KNB; ++k) wacc += W.E[lane][k] * W.ci[k];
  wacc *= (1.0f / KNB);                 // w[j] = (1/K) sum_k E[j,k]/c[k]
  W.wv[lane] = wacc;
  W.tv[lane] = wacc * W.ri[lane];       // t[j] = w[j]/r[j]
  __syncthreads();

  float uacc = 0.f;
  for (int j = 0; j < KNB; ++j) uacc += W.E[j][lane] * W.tv[j];  // u = E^T t
  W.uv[lane] = uacc;
  __syncthreads();

  // ---------- fused reductions: Hm = [mean_k Qm, w^T Dm, u^T Qm] ----------
  const int d0 = lane * 4;
  float m1[4] = {}, m2[4] = {}, m3[4] = {};
  for (int j = 0; j < KNB; ++j) {
    const float wj = W.wv[j];
    const float uj = W.uv[j];
    #pragma unroll
    for (int q = 0; q < 4; ++q) {
      const float qv = (float)W.Qm[j][d0 + q];
      const float dv = (float)W.Dm[j][d0 + q];
      m1[q] += qv;
      m2[q] = fmaf(wj, dv, m2[q]);
      m3[q] = fmaf(uj, qv, m3[q]);
    }
  }
  #pragma unroll
  for (int q = 0; q < 4; ++q) {
    W.Hm[d0 + q]          = m1[q] * (1.0f / KNB);
    W.Hm[DF + d0 + q]     = m2[q];
    W.Hm[2 * DF + d0 + q] = m3[q];
  }
  __syncthreads();

  // ---------- avg-pool-3 over the flat 384 vector + residual add ----------
  float4 fv = ((const float4*)(feat + (size_t)node * DF))[lane];
  float o[4];
  #pragma unroll
  for (int q = 0; q < 4; ++q) {
    const int c = d0 + q;
    o[q] = (W.Hm[3 * c] + W.Hm[3 * c + 1] + W.Hm[3 * c + 2]) * (1.0f / 3.0f);
  }
  float4 ov;
  ov.x = fv.x + o[0]; ov.y = fv.y + o[1];
  ov.z = fv.z + o[2]; ov.w = fv.w + o[3];
  ((float4*)(out + (size_t)node * DF))[lane] = ov;
}

extern "C" void kernel_launch(void* const* d_in, const int* in_sizes, int n_in,
                              void* d_out, int out_size, void* d_ws, size_t ws_size,
                              hipStream_t stream) {
  const float* feat = (const float*)d_in[0];
  const int*   sim  = (const int*)d_in[1];
  const int*   cor  = (const int*)d_in[2];
  float* out = (float*)d_out;
  dim3 grid(N_NODES / WPB);
  dim3 block(WPB * 32);
  hipLaunchKernelGGL(coatt_kernel, grid, block, 0, stream, feat, sim, cor, out);
}
